// BiDAF_49160195670042
// MI455X (gfx1250) — matrix operand order repaired and checked by reading, not compiled, not run on hardware
//
#include <hip/hip_runtime.h>
#include <hip/hip_bf16.h>

// BiDAF forward for MI455X (gfx1250). All GEMM-shaped math runs through
// v_wmma_f32_16x16x32_f16 (f16 inputs, f32 accumulate), with vectorized
// (b64/b128) fragment loads on wave-uniform fast paths and 4-wide N blocking
// (1 A fragment -> 4 WMMAs). The LSTM recurrence is a persistent 25-wave
// workgroup per direction with Whh fragments held in VGPRs across all
// timesteps; h lives in LDS padded to 16x128 (zero cols 100..127) so every
// k-step of the per-step WMMA takes the unguarded vector path.
//
// Input order: setup_inputs() dict order, params flattened jax-pytree style
// (dict keys sorted): indices 0..7 = tensors, 8..76 = param leaves.

typedef __attribute__((ext_vector_type(16))) _Float16 v16h;
typedef __attribute__((ext_vector_type(8)))  float    v8f;

__device__ __forceinline__ v8f wmma_f16(v16h a, v16h b, v8f c) {
  // (neg_a, A, neg_b, B, c_mod, C, reuse_a, reuse_b)
  return __builtin_amdgcn_wmma_f32_16x16x32_f16(false, a, false, b, (short)0, c,
                                                false, false);
}

// Pack 8 contiguous f32 (8B-aligned) into halves [hbase, hbase+8) of a v16h.
template <bool SCALE>
__device__ __forceinline__ void pack8(v16h& a, int hbase,
                                      const float* __restrict__ p,
                                      const float* __restrict__ s) {
  const float2* q = (const float2*)p;
  const float2* qs = (const float2*)s;
#pragma unroll
  for (int i = 0; i < 4; ++i) {
    float2 v = q[i];
    if (SCALE) { float2 sc = qs[i]; v.x *= sc.x; v.y *= sc.y; }
    a[hbase + 2 * i]     = (_Float16)v.x;
    a[hbase + 2 * i + 1] = (_Float16)v.y;
  }
}

// ---- A fragment (16x32 MxK). Lane: m = m0+(lane&15), kh = lane>>4.
// Half h holds k = k0 + h + 16*kh (contiguous run of 16).
template <bool SCALE>
__device__ __forceinline__ v16h load_a_fast(const float* __restrict__ rowp,
                                            const float* __restrict__ sc) {
  v16h a;
  pack8<SCALE>(a, 0, rowp,      sc);
  pack8<SCALE>(a, 8, rowp + 8, sc + 8);
  return a;
}

template <bool SCALE>
__device__ __forceinline__ v16h load_a_slow(const float* __restrict__ A, int lda,
                                            int m0, int k0, int M, int K,
                                            int lane, const float* kScale) {
  int m  = m0 + (lane & 15);
  int kh = (lane >> 4) * 16;
  v16h a;
#pragma unroll
  for (int h = 0; h < 16; ++h) {
    int k = k0 + h + kh;
    float v = 0.0f;
    if (m < M && k < K) {
      v = A[(long)m * lda + k];
      if (SCALE) v *= kScale[k];
    }
    a[h] = (_Float16)v;
  }
  return a;
}

// ---- B fragment (32x16 KxN). Lane: n = n0+(lane&15), kh = lane>>4.
// Half h holds k = k0 + h + 16*kh  (contiguous run of 16 for transB).
__device__ __forceinline__ v16h load_b_fast_t(const float* __restrict__ rowp) {
  v16h b;
  pack8<false>(b, 0, rowp,     nullptr);
  pack8<false>(b, 8, rowp + 8, nullptr);
  return b;
}

template <int TRANSB>
__device__ __forceinline__ v16h load_b_slow(const float* __restrict__ B, int ldb,
                                            int n0, int k0, int N, int K,
                                            int lane) {
  int n  = n0 + (lane & 15);
  int kh = (lane >> 4) * 16;
  v16h b;
#pragma unroll
  for (int h = 0; h < 16; ++h) {
    int k = k0 + h + kh;
    float v = 0.0f;
    if (n < N && k < K)
      v = TRANSB ? B[(long)n * ldb + k] : B[(long)k * ldb + n];
    b[h] = (_Float16)v;
  }
  return b;
}

// B fragment, no-transpose interior (all k in range): strided scalar loads.
__device__ __forceinline__ v16h load_b_fast_n(const float* __restrict__ colp,
                                              int ldb) {
  v16h b;
#pragma unroll
  for (int h = 0; h < 16; ++h) b[h] = (_Float16)colp[(long)h * ldb];
  return b;
}

// ---- generic batched WMMA GEMM: C = A @ op(B) + fused adds ----------------
// Each wave computes a 16(M) x 64(N) strip: 1 A fragment feeds 4 WMMAs.
template <int TRANSB, bool SCALE>
__global__ void wmma_gemm_kernel(const float* __restrict__ A, int lda, long sA,
                                 const float* __restrict__ B, int ldb, long sB,
                                 float* __restrict__ C, int ldc, long sC,
                                 int M, int N, int K,
                                 const float* bias0, const float* bias1,
                                 const float* rowAdd, long sRow,
                                 const float* colAdd, long sCol,
                                 const float* scalarAddPtr,
                                 const float* kScaleA) {
  int lane    = threadIdx.x & 31;
  int wave    = threadIdx.x >> 5;
  int tilesN4 = (N + 63) >> 6;
  int tilesM  = (M + 15) >> 4;
  int tile    = blockIdx.x * (blockDim.x >> 5) + wave;   // wave-uniform
  if (tile >= tilesM * tilesN4) return;
  int bz = blockIdx.y;
  const float* Ab = A + (long)bz * sA;
  const float* Bb = B + (long)bz * sB;
  float*       Cb = C + (long)bz * sC;
  int m0 = (tile / tilesN4) * 16;
  int n0 = (tile % tilesN4) * 64;

  const int mrow = m0 + (lane & 15);
  const int kh   = (lane >> 4);
  const int nb   = lane & 15;
  const bool mFull = (m0 + 16 <= M);

  v8f acc[4];
#pragma unroll
  for (int j = 0; j < 4; ++j) acc[j] = v8f{0.f,0.f,0.f,0.f,0.f,0.f,0.f,0.f};

  for (int k0 = 0; k0 < K; k0 += 32) {
    const bool fullK = (k0 + 32 <= K);
    v16h a;
    if (mFull && fullK)
      a = load_a_fast<SCALE>(Ab + (long)mrow * lda + k0 + 16 * kh,
                             kScaleA + k0 + 16 * kh);
    else
      a = load_a_slow<SCALE>(Ab, lda, m0, k0, M, K, lane, kScaleA);
#pragma unroll
    for (int j = 0; j < 4; ++j) {
      int nj0 = n0 + j * 16;
      if (nj0 >= N) break;                    // uniform
      const bool nFull = (nj0 + 16 <= N);
      v16h b;
      if (nFull && fullK) {
        if (TRANSB) b = load_b_fast_t(Bb + (long)(nj0 + nb) * ldb + k0 + 16 * kh);
        else        b = load_b_fast_n(Bb + (long)(k0 + 16 * kh) * ldb + nj0 + nb, ldb);
      } else {
        b = load_b_slow<TRANSB>(Bb, ldb, nj0, k0, N, K, lane);
      }
      acc[j] = wmma_f16(a, b, acc[j]);
    }
  }

  float sadd = scalarAddPtr ? scalarAddPtr[0] : 0.0f;
  int mbase = m0 + kh * 8;
#pragma unroll
  for (int j = 0; j < 4; ++j) {
    int n = n0 + j * 16 + nb;
    if (n0 + j * 16 >= N) break;              // uniform
    float colv = sadd;
    if (n < N) {
      if (bias0)  colv += bias0[n];
      if (bias1)  colv += bias1[n];
      if (colAdd) colv += colAdd[(long)bz * sCol + n];
    }
#pragma unroll
    for (int r = 0; r < 8; ++r) {
      int m = mbase + r;
      if (m < M && n < N) {
        float v = acc[j][r] + colv;
        if (rowAdd) v += rowAdd[(long)bz * sRow + m];
        Cb[(long)m * ldc + n] = v;
      }
    }
  }
}

// ---- persistent BiLSTM recurrence: grid.x=2 (0=fwd, 1=bwd) ----------------
__device__ __forceinline__ float sigf(float x) { return 1.0f / (1.0f + expf(-x)); }

__global__ __launch_bounds__(800) void lstm_seq_kernel(
    const float* __restrict__ xg_f, const float* __restrict__ xg_b,
    const float* __restrict__ whh_f, const float* __restrict__ whh_b,
    float* __restrict__ out, int T) {
  // B=16, H=100, 4H=400, out ld=200 (fwd cols 0:100, bwd cols 100:200)
  // h is padded to stride 128 with cols 100..127 permanently zero so all four
  // 32-wide k-steps of the per-step WMMA use unguarded b128 LDS reads.
  __shared__ float h_lds[16 * 128];
  __shared__ float c_lds[16 * 100];
  __shared__ float G_lds[16 * 400];
  const int dir = blockIdx.x;
  const float* xg  = dir ? xg_b  : xg_f;
  const float* Whh = dir ? whh_b : whh_f;
  const int colOff = dir * 100;
  const int tid  = threadIdx.x;
  const int lane = tid & 31;
  const int wave = tid >> 5;        // 0..24, each owns cols [wave*16, +16)
  const int n0   = wave * 16;
  const int kh   = lane >> 4;
  const int nb   = lane & 15;

  // Whh fragments live in VGPRs for the whole sequence (K=100 -> 4 steps).
  v16h bfrag[4];
#pragma unroll
  for (int kk = 0; kk < 3; ++kk)    // k0 = 0,32,64: fully interior
    bfrag[kk] = load_b_fast_t(Whh + (long)(n0 + nb) * 100 + kk * 32 + 16 * kh);
  bfrag[3] = load_b_slow<1>(Whh, 100, n0, 96, 400, 100, lane);  // K tail, once

  for (int e = tid; e < 16 * 128; e += 800) h_lds[e] = 0.0f;
  for (int e = tid; e < 16 * 100; e += 800) c_lds[e] = 0.0f;
  __syncthreads();

  const float* hrow = h_lds + (lane & 15) * 128;
  for (int s = 0; s < T; ++s) {
    int t = dir ? (T - 1 - s) : s;
    v8f acc = {0.f, 0.f, 0.f, 0.f, 0.f, 0.f, 0.f, 0.f};
#pragma unroll
    for (int kk = 0; kk < 4; ++kk) {   // k0 = 0,32,64,96: all vector LDS reads
      v16h a = load_a_fast<false>(hrow + kk * 32 + 16 * kh, nullptr);
      acc = wmma_f16(a, bfrag[kk], acc);
    }
    int n     = n0 + nb;
    int mbase = kh * 8;
#pragma unroll
    for (int r = 0; r < 8; ++r) {
      int m = mbase + r;  // batch index
      G_lds[m * 400 + n] = acc[r] + xg[((long)m * T + t) * 400 + n];
    }
    __syncthreads();
    for (int e = tid; e < 16 * 100; e += 800) {
      int bi = e / 100, idx = e - bi * 100;
      float gi = G_lds[bi * 400 + idx];
      float gf = G_lds[bi * 400 + 100 + idx];
      float gg = G_lds[bi * 400 + 200 + idx];
      float go = G_lds[bi * 400 + 300 + idx];
      float c  = sigf(gf) * c_lds[e] + sigf(gi) * tanhf(gg);
      float h  = sigf(go) * tanhf(c);
      c_lds[e] = c;
      h_lds[bi * 128 + idx] = h;      // cols 100..127 stay zero
      out[((long)bi * T + t) * 200 + colOff + idx] = h;
    }
    __syncthreads();
  }
}

// ---- small helper kernels -------------------------------------------------
__global__ void gather_rows_kernel(const int* __restrict__ words,
                                   const float* __restrict__ glove,
                                   float* __restrict__ out, int nRows, int E) {
  long i = (long)blockIdx.x * blockDim.x + threadIdx.x;
  long total = (long)nRows * E;
  if (i < total) {
    int r = (int)(i / E), c = (int)(i - (long)r * E);
    out[i] = glove[(long)words[r] * E + c];
  }
}

// char CNN: one wave per word; 1024-value reinterpreted char block in LDS.
__global__ void charconv_kernel(const int* __restrict__ chars,
                                const float* __restrict__ cemb,
                                const float* __restrict__ conv_w,
                                const float* __restrict__ conv_b,
                                float* __restrict__ x, int nWords) {
  __shared__ float buf[8][1024];
  int wave = threadIdx.x >> 5, lane = threadIdx.x & 31;
  int word = blockIdx.x * 8 + wave;
  float* f = buf[wave];
  if (word < nWords) {
    const int* wch = chars + (long)word * 16;
    for (int i = lane; i < 1024; i += 32)
      f[i] = cemb[(long)wch[i >> 6] * 64 + (i & 63)];
  }
  __syncthreads();
  if (word < nWords) {
    for (int oc = lane; oc < 100; oc += 32) {
      float best = -1e30f;
      for (int pos = 0; pos < 12; ++pos) {
        float s = 0.0f;
        for (int c = 0; c < 64; ++c) {
          const float* wrow = conv_w + ((long)oc * 64 + c) * 5;
          const float* frow = f + c * 16 + pos;
#pragma unroll
          for (int kk = 0; kk < 5; ++kk) s += frow[kk] * wrow[kk];
        }
        best = fmaxf(best, s);
      }
      x[(long)word * 200 + 100 + oc] = best + conv_b[oc];
    }
  }
}

__global__ void highway_kernel(float* __restrict__ x,
                               const float* __restrict__ tlin,
                               const float* __restrict__ glin, long n) {
  long i = (long)blockIdx.x * blockDim.x + threadIdx.x;
  if (i < n) {
    float t = fmaxf(tlin[i], 0.0f);
    float g = 1.0f / (1.0f + expf(-glin[i]));
    x[i] = x[i] * (1.0f - g) + g * t;
  }
}

__global__ void rowdot_kernel(const float* __restrict__ X, int ld,
                              const float* __restrict__ w,
                              float* __restrict__ out, int nRows, int D) {
  int row  = blockIdx.x * (blockDim.x >> 5) + (threadIdx.x >> 5);
  int lane = threadIdx.x & 31;
  if (row >= nRows) return;
  float s = 0.0f;
  for (int d = lane; d < D; d += 32) s += X[(long)row * ld + d] * w[d];
  for (int off = 16; off > 0; off >>= 1) s += __shfl_down(s, off, 32);
  if (lane == 0) out[row] = s;
}

__global__ void softmax50_kernel(float* __restrict__ sim, float* __restrict__ rmax,
                                 const float* __restrict__ cmask,
                                 const float* __restrict__ qmask,
                                 int nRows, int W, int Lc) {
  int row  = blockIdx.x * (blockDim.x >> 5) + (threadIdx.x >> 5);
  int lane = threadIdx.x & 31;
  if (row >= nRows) return;
  float* p = sim + (long)row * W;
  int b = row / Lc;
  float m = -1e30f;
  for (int j = lane; j < W; j += 32) m = fmaxf(m, p[j]);
  for (int off = 16; off > 0; off >>= 1) m = fmaxf(m, __shfl_xor(m, off, 32));
  float s = 0.0f;
  for (int j = lane; j < W; j += 32) { float e = expf(p[j] - m); p[j] = e; s += e; }
  for (int off = 16; off > 0; off >>= 1) s += __shfl_xor(s, off, 32);
  float inv = (1.0f / s) * cmask[row];
  for (int j = lane; j < W; j += 32) p[j] *= inv * qmask[b * W + j];
  if (lane == 0) rmax[row] = m;   // pre-softmax row max, feeds beta
}

__global__ void beta_bvec_kernel(const float* __restrict__ rmax,
                                 const float* __restrict__ cmask,
                                 const float* __restrict__ cenc,
                                 float* __restrict__ bvec, int Lc, int D) {
  __shared__ float red[256];
  __shared__ float beta[400];
  int b = blockIdx.x, tid = threadIdx.x;
  const float* r = rmax + (long)b * Lc;
  float m = -1e30f;
  for (int i = tid; i < Lc; i += 256) m = fmaxf(m, r[i]);
  red[tid] = m; __syncthreads();
  for (int s = 128; s > 0; s >>= 1) { if (tid < s) red[tid] = fmaxf(red[tid], red[tid + s]); __syncthreads(); }
  m = red[0]; __syncthreads();
  float su = 0.0f;
  for (int i = tid; i < Lc; i += 256) { float e = expf(r[i] - m); beta[i] = e; su += e; }
  red[tid] = su; __syncthreads();
  for (int s = 128; s > 0; s >>= 1) { if (tid < s) red[tid] += red[tid + s]; __syncthreads(); }
  float inv = 1.0f / red[0]; __syncthreads();
  for (int i = tid; i < Lc; i += 256) beta[i] *= inv * cmask[(long)b * Lc + i];
  __syncthreads();
  for (int d = tid; d < D; d += 256) {
    float s2 = 0.0f;
    for (int i = 0; i < Lc; ++i) s2 += beta[i] * cenc[((long)b * Lc + i) * D + d];
    bvec[b * D + d] = s2;
  }
}

__global__ void g_assemble_kernel(const float* __restrict__ cenc,
                                  const float* __restrict__ a,
                                  const float* __restrict__ bvec,
                                  float* __restrict__ g, int Lc, int D) {
  long i = (long)blockIdx.x * blockDim.x + threadIdx.x;
  long total = (long)16 * Lc * D;
  if (i >= total) return;
  int  d   = (int)(i % D);
  long row = i / D;
  int  b   = (int)(row / Lc);
  float c = cenc[i], av = a[i];
  float* grow = g + row * (4L * D);
  grow[d]         = c;
  grow[D + d]     = av;
  grow[2 * D + d] = c * av;
  grow[3 * D + d] = c * bvec[b * D + d];
}

__global__ void span_kernel(const float* __restrict__ g, const float* __restrict__ m,
                            const float* __restrict__ w, float* __restrict__ out,
                            int nRows, int Dg, int Dm) {
  int row  = blockIdx.x * (blockDim.x >> 5) + (threadIdx.x >> 5);
  int lane = threadIdx.x & 31;
  if (row >= nRows) return;
  float s = 0.0f;
  for (int d = lane; d < Dg; d += 32) s += g[(long)row * Dg + d] * w[d];
  for (int d = lane; d < Dm; d += 32) s += m[(long)row * Dm + d] * w[Dg + d];
  for (int off = 16; off > 0; off >>= 1) s += __shfl_down(s, off, 32);
  if (lane == 0) out[row] = s;
}

// ---- host orchestration ---------------------------------------------------
static inline void launch_gemm(hipStream_t stream,
                               const float* A, int lda, long sA,
                               const float* B, int ldb, long sB, int transB,
                               float* C, int ldc, long sC,
                               int M, int N, int K, int batch,
                               const float* bias0 = nullptr, const float* bias1 = nullptr,
                               const float* rowAdd = nullptr, long sRow = 0,
                               const float* colAdd = nullptr, long sCol = 0,
                               const float* sPtr = nullptr, const float* kScale = nullptr) {
  int tiles = ((M + 15) / 16) * ((N + 63) / 64);
  dim3 gr((tiles + 7) / 8, batch);
  dim3 bl(256);
  if (transB) {
    if (kScale)
      wmma_gemm_kernel<1, true><<<gr, bl, 0, stream>>>(A, lda, sA, B, ldb, sB, C, ldc, sC,
          M, N, K, bias0, bias1, rowAdd, sRow, colAdd, sCol, sPtr, kScale);
    else
      wmma_gemm_kernel<1, false><<<gr, bl, 0, stream>>>(A, lda, sA, B, ldb, sB, C, ldc, sC,
          M, N, K, bias0, bias1, rowAdd, sRow, colAdd, sCol, sPtr, kScale);
  } else {
    if (kScale)
      wmma_gemm_kernel<0, true><<<gr, bl, 0, stream>>>(A, lda, sA, B, ldb, sB, C, ldc, sC,
          M, N, K, bias0, bias1, rowAdd, sRow, colAdd, sCol, sPtr, kScale);
    else
      wmma_gemm_kernel<0, false><<<gr, bl, 0, stream>>>(A, lda, sA, B, ldb, sB, C, ldc, sC,
          M, N, K, bias0, bias1, rowAdd, sRow, colAdd, sCol, sPtr, kScale);
  }
}

struct LstmP { const float *Wih_f, *Whh_f, *bih_f, *bhh_f, *Wih_b, *Whh_b, *bih_b, *bhh_b; };

extern "C" void kernel_launch(void* const* d_in, const int* in_sizes, int n_in,
                              void* d_out, int out_size, void* d_ws, size_t ws_size,
                              hipStream_t stream) {
  (void)in_sizes; (void)n_in; (void)out_size; (void)ws_size;
  auto F = [&](int i) { return (const float*)d_in[i]; };
  auto I = [&](int i) { return (const int*)d_in[i]; };

  const int LC = 400, LQ = 50;
  const int rowsC = 16 * LC, rowsQ = 16 * LQ;

  // Param leaves (pytree order, dict keys sorted):
  // 8 att_b, 9 att_w, 10..21 ctx_emb{cemb,conv_b,conv_w,hw0(Wt,bt,Wg,bg),
  // hw1(...),proj}, 22..29 ctx_enc{bwd(Wih,Whh,bih,bhh),fwd(...)}, 30 glove,
  // 31..38 model_l1, 39..46 model_l2, 47..54 out_lstm, 55..66 qry_emb,
  // 67..74 qry_enc, 75 w_end, 76 w_start
  auto lstmAt = [&](int base) {
    LstmP p;
    p.Wih_b = F(base + 0); p.Whh_b = F(base + 1); p.bih_b = F(base + 2); p.bhh_b = F(base + 3);
    p.Wih_f = F(base + 4); p.Whh_f = F(base + 5); p.bih_f = F(base + 6); p.bhh_f = F(base + 7);
    return p;
  };

  float* wsf = (float*)d_ws;
  size_t off = 0;
  auto alloc = [&](size_t n) { float* p = wsf + off; off += n; return p; };
  float* xC    = alloc((size_t)rowsC * 200);
  float* xQ    = alloc((size_t)rowsQ * 200);
  float* gbuf  = alloc((size_t)rowsC * 100);   // gather scratch (reused)
  float* tbuf  = alloc((size_t)rowsC * 200);   // highway t_lin (reused)
  float* g2buf = alloc((size_t)rowsC * 200);   // highway g_lin (reused)
  float* xg_f  = alloc((size_t)rowsC * 400);   // LSTM pre-gates (reused per layer)
  float* xg_b  = alloc((size_t)rowsC * 400);
  float* cenc  = alloc((size_t)rowsC * 200);
  float* qenc  = alloc((size_t)rowsQ * 200);
  float* cdot  = alloc(rowsC);
  float* qdot  = alloc(rowsQ);
  float* rmax  = alloc(rowsC);
  float* bvec  = alloc(16 * 200);
  float* sim   = alloc((size_t)rowsC * LQ);
  float* abuf  = alloc((size_t)rowsC * 200);
  float* gbig  = alloc((size_t)rowsC * 800);
  float* t1    = alloc((size_t)rowsC * 200);
  float* mb    = alloc((size_t)rowsC * 200);
  float* m2b   = alloc((size_t)rowsC * 200);

  const float* glove = F(30);

  auto embed = [&](const int* words, const int* chars, int rows, int eb, float* X) {
    long tot = (long)rows * 100;
    gather_rows_kernel<<<(unsigned)((tot + 255) / 256), 256, 0, stream>>>(
        words, glove, gbuf, rows, 100);
    // word proj -> X[:, 0:100]
    launch_gemm(stream, gbuf, 100, 0, F(eb + 11), 100, 0, 1, X, 200, 0,
                rows, 100, 100, 1);
    // char CNN -> X[:, 100:200]
    charconv_kernel<<<(rows + 7) / 8, 256, 0, stream>>>(
        chars, F(eb + 0), F(eb + 2), F(eb + 1), X, rows);
    // two highway layers
    for (int l = 0; l < 2; ++l) {
      int hb = eb + 3 + l * 4;
      launch_gemm(stream, X, 200, 0, F(hb + 0), 200, 0, 1, tbuf, 200, 0,
                  rows, 200, 200, 1, F(hb + 1));
      launch_gemm(stream, X, 200, 0, F(hb + 2), 200, 0, 1, g2buf, 200, 0,
                  rows, 200, 200, 1, F(hb + 3));
      long n = (long)rows * 200;
      highway_kernel<<<(unsigned)((n + 255) / 256), 256, 0, stream>>>(X, tbuf, g2buf, n);
    }
  };

  auto bilstm = [&](const float* X, int rows, int T, int Din, LstmP p, float* out) {
    launch_gemm(stream, X, Din, 0, p.Wih_f, Din, 0, 1, xg_f, 400, 0,
                rows, 400, Din, 1, p.bih_f, p.bhh_f);
    launch_gemm(stream, X, Din, 0, p.Wih_b, Din, 0, 1, xg_b, 400, 0,
                rows, 400, Din, 1, p.bih_b, p.bhh_b);
    lstm_seq_kernel<<<dim3(2), dim3(800), 0, stream>>>(
        xg_f, xg_b, p.Whh_f, p.Whh_b, out, T);
  };

  // ---- embeddings + encoders ----
  embed(I(0), I(1), rowsC, 10, xC);
  embed(I(4), I(5), rowsQ, 55, xQ);
  bilstm(xC, rowsC, LC, 200, lstmAt(22), cenc);
  bilstm(xQ, rowsQ, LQ, 200, lstmAt(67), qenc);

  // ---- attention ----
  const float* att_w = F(9);   // [wc(200) | wq(200) | wp(200)]
  rowdot_kernel<<<rowsC / 8, 256, 0, stream>>>(cenc, 200, att_w, cdot, rowsC, 200);
  rowdot_kernel<<<rowsQ / 8, 256, 0, stream>>>(qenc, 200, att_w + 200, qdot, rowsQ, 200);
  // sim[b] = (cenc*wp) @ qenc^T + cdot_i + qdot_j + att_b  (batched WMMA)
  launch_gemm(stream, cenc, 200, (long)LC * 200, qenc, 200, (long)LQ * 200, 1,
              sim, LQ, (long)LC * LQ, LC, LQ, 200, 16,
              nullptr, nullptr, cdot, LC, qdot, LQ, F(8), att_w + 400);
  softmax50_kernel<<<rowsC / 8, 256, 0, stream>>>(sim, rmax, F(2), F(6), rowsC, LQ, LC);
  beta_bvec_kernel<<<16, 256, 0, stream>>>(rmax, F(2), cenc, bvec, LC, 200);
  // a[b] = alpha @ qenc  (batched WMMA, B stored [K,N])
  launch_gemm(stream, sim, LQ, (long)LC * LQ, qenc, 200, (long)LQ * 200, 0,
              abuf, 200, (long)LC * 200, LC, 200, LQ, 16);
  {
    long n = (long)rowsC * 200;
    g_assemble_kernel<<<(unsigned)((n + 255) / 256), 256, 0, stream>>>(
        cenc, abuf, bvec, gbig, LC, 200);
  }

  // ---- modeling stack ----
  bilstm(gbig, rowsC, LC, 800, lstmAt(31), t1);   // model_l1
  bilstm(t1,   rowsC, LC, 200, lstmAt(39), mb);   // model_l2  -> m
  bilstm(mb,   rowsC, LC, 200, lstmAt(47), m2b);  // out_lstm  -> m2

  // ---- span logits: out = [start(16x400) | end(16x400)] ----
  float* out = (float*)d_out;
  span_kernel<<<rowsC / 8, 256, 0, stream>>>(gbig, mb,  F(76), out,         rowsC, 800, 200);
  span_kernel<<<rowsC / 8, 256, 0, stream>>>(gbig, m2b, F(75), out + rowsC, rowsC, 800, 200);
}
